// MultiHeadAttention_64622077935980
// MI455X (gfx1250) — matrix operand order repaired and checked by reading, not compiled, not run on hardware
//
#include <hip/hip_runtime.h>
#include <hip/hip_bf16.h>
#include <stdint.h>

// ---------------------------------------------------------------------------
// MHA forward on gfx1250 (CDNA5): bf16 WMMA GEMMs + flash-attention softmax.
// - All matmuls: v_wmma_f32_16x16x32_bf16 (wave32, f32 accumulate)
// - GEMMs two-phase ping-pong pipelined (no cur/nxt register copies)
// - Flash K/V tiles staged to LDS by the Tensor Data Mover (double-buffered,
//   synced with s_wait_tensorcnt), shared by all 8 waves of the block.
// - P-tile layout bounce uses wave-local s_wait_dscnt (LDS in-order per wave)
//   instead of block barriers.
// ---------------------------------------------------------------------------

#define B_  2
#define S_  2048
#define D_  1024
#define H_  16
#define DH_ 64

typedef __attribute__((ext_vector_type(16))) __bf16       v16bf;
typedef __attribute__((ext_vector_type(8)))  float        v8f;
typedef __attribute__((ext_vector_type(4)))  unsigned int v4u;
typedef __attribute__((ext_vector_type(8)))  int          v8i_t;
typedef __attribute__((ext_vector_type(4)))  int          v4i_t;

#if __has_builtin(__builtin_amdgcn_tensor_load_to_lds) && \
    __has_builtin(__builtin_amdgcn_s_wait_tensorcnt)
#define USE_TDM 1
#else
#define USE_TDM 0
#endif

__device__ __forceinline__ v8f wmma_bf16(v16bf a, v16bf b, v8f c) {
  return __builtin_amdgcn_wmma_f32_16x16x32_bf16(
      false, a, false, b, (short)0, c, false, false);
}

// Wave-local LDS ordering: stores->loads within one wave only need DScnt==0
// (LDS ops from a wave complete in order; lanes are lockstep in wave32).
__device__ __forceinline__ void wave_lds_fence() {
  asm volatile("s_wait_dscnt 0x0" ::: "memory");
}

// A-matrix (16x32 bf16) per-lane element -> K index (ISA 7.12.2 16-bit A table)
__device__ __forceinline__ int kmapA(int i, int half) {
  return ((i & 8) << 1) + (i & 7) + half * 8;
}
__device__ __forceinline__ v16bf load_a(const __bf16* rowptr, int k0, int half) {
  v16bf a;
#pragma unroll
  for (int i = 0; i < 16; ++i) a[i] = rowptr[k0 + kmapA(i, half)];
  return a;
}
// B-matrix (32x16): lane column n = lane&15; elements K = k0 + kmapA(i, half).
__device__ __forceinline__ v16bf load_b(const __bf16* rowptr, int k0, int half) {
  v16bf b;
#pragma unroll
  for (int i = 0; i < 16; ++i) b[i] = rowptr[k0 + kmapA(i, half)];
  return b;
}

__device__ __forceinline__ float hmax16(float v) {
  v = fmaxf(v, __shfl_xor(v, 1));
  v = fmaxf(v, __shfl_xor(v, 2));
  v = fmaxf(v, __shfl_xor(v, 4));
  v = fmaxf(v, __shfl_xor(v, 8));
  return v;
}
__device__ __forceinline__ float hsum16(float v) {
  v += __shfl_xor(v, 1);
  v += __shfl_xor(v, 2);
  v += __shfl_xor(v, 4);
  v += __shfl_xor(v, 8);
  return v;
}

// ---------------------------------------------------------------------------
// Tensor Data Mover: build D# descriptor (ISA 08_async_tensor §8.3-8.5) and
// issue a 1D/2D bf16 tile load into LDS.  tile_d1==0 -> 1D tile.
#if USE_TDM
__device__ __forceinline__ void tdm_load(uint32_t lds_addr, const void* gptr,
                                         uint32_t tensor_d0, uint32_t tensor_d1,
                                         uint32_t tile_d0, uint32_t tile_d1,
                                         uint64_t stride0) {
  uint64_t ga = (uint64_t)(uintptr_t)gptr;
  union { v4u v; uint32_t d[4]; } g0;
  g0.d[0] = 1u;                                        // count=1, user mode
  g0.d[1] = lds_addr;                                  // lds_addr
  g0.d[2] = (uint32_t)ga;                              // global_addr[31:0]
  g0.d[3] = (uint32_t)((ga >> 32) & 0x01FFFFFFu) | (2u << 30);  // addr hi, type=2
  union { v8i_t v; uint64_t q[4]; } g1;
  const uint64_t dsz = 1;                              // 2-byte elements
  g1.q[0] = (dsz << 16)                                // data_size
          | ((uint64_t)(tensor_d0 & 0xFFFFu) << 48);   // tensor_dim0[15:0]
  g1.q[1] = (uint64_t)(tensor_d0 >> 16)                // tensor_dim0[31:16]
          | ((uint64_t)tensor_d1 << 16)                // tensor_dim1
          | ((uint64_t)(tile_d0 & 0xFFFFu) << 48);     // tile_dim0
  g1.q[2] = (uint64_t)(tile_d1 & 0xFFFFu)              // tile_dim1 (0 = 1D)
          | ((stride0 & 0xFFFFFFFFull) << 32);         // dim0_stride[31:0]
  g1.q[3] = ((stride0 >> 32) & 0xFFFFull);             // dim0_stride[47:32]
  v4i_t z4 = (v4i_t)0;
#if defined(__clang_major__) && (__clang_major__ >= 23)
  v8i_t z8 = (v8i_t)0;
  __builtin_amdgcn_tensor_load_to_lds(g0.v, g1.v, z4, z4, z8, 0);
#else
  __builtin_amdgcn_tensor_load_to_lds(g0.v, g1.v, z4, z4, 0);
#endif
}
__device__ __forceinline__ uint32_t lds_off(const void* p) {
  return (uint32_t)(uintptr_t)p;   // LDS flat address: low 32 bits = LDS offset
}
#endif

// ---------------------------------------------------------------------------
// GEMM fragment helpers: one wave computes 32 rows x 64 cols per k-step.
struct Frag { v16bf A0, A1, B[4]; };

__device__ __forceinline__ void load_frag(Frag& f,
                                          const __bf16* xr0, const __bf16* xr1,
                                          const __bf16* wbase, int m, int k0,
                                          int half) {
  f.A0 = load_a(xr0, k0, half);
  f.A1 = load_a(xr1, k0, half);
#pragma unroll
  for (int t = 0; t < 4; ++t)
    f.B[t] = load_b(wbase + (size_t)(t * 16 + m) * D_, k0, half);
}

__device__ __forceinline__ void gemm_step(const Frag& f, v8f acc0[4], v8f acc1[4]) {
#pragma unroll
  for (int t = 0; t < 4; ++t) {
    acc0[t] = wmma_bf16(f.A0, f.B[t], acc0[t]);
    acc1[t] = wmma_bf16(f.A1, f.B[t], acc1[t]);
  }
}

// ---------------------------------------------------------------------------
// K0a: fp32 -> bf16 copy
__global__ void k_cvt(const float* __restrict__ in, __bf16* __restrict__ out, int n) {
  for (int i = blockIdx.x * blockDim.x + threadIdx.x; i < n;
       i += gridDim.x * blockDim.x)
    out[i] = (__bf16)in[i];
}

// K0b: fp32 [batch][rows][cols] -> bf16 [batch][cols][rows] (transpose)
__global__ void k_cvt_tr(const float* __restrict__ in, __bf16* __restrict__ out,
                         int batch, int rows, int cols) {
  int total = batch * rows * cols;
  for (int i = blockIdx.x * blockDim.x + threadIdx.x; i < total;
       i += gridDim.x * blockDim.x) {
    int b  = i / (rows * cols);
    int rc = i % (rows * cols);
    int r = rc / cols, c = rc % cols;
    out[b * rows * cols + c * rows + r] = (__bf16)in[i];
  }
}

// ---------------------------------------------------------------------------
// K1: QKV projection.  One wave -> 32 rows x 64 cols of one head's Q/K/V.
// Two-phase ping-pong pipeline (16 WMMAs / iteration, no register copies).
// Weights pre-transposed [H][DH][D]; V stored transposed [b,h,dh,s].
__global__ void k_qkv(const __bf16* __restrict__ xbf,
                      const __bf16* __restrict__ wqt,
                      const __bf16* __restrict__ wkt,
                      const __bf16* __restrict__ wvt,
                      const float* __restrict__ bq,
                      const float* __restrict__ bk,
                      const float* __restrict__ bv,
                      __bf16* __restrict__ qo,
                      __bf16* __restrict__ ko,
                      __bf16* __restrict__ vto) {
  const int w    = threadIdx.x >> 5;
  const int lane = threadIdx.x & 31;
  const int m    = lane & 15;
  const int half = lane >> 4;

  const int which = blockIdx.y / (B_ * H_);
  const int bh    = blockIdx.y % (B_ * H_);
  const int b     = bh / H_;
  const int h     = bh % H_;

  const __bf16* wt  = (which == 0) ? wqt : (which == 1) ? wkt : wvt;
  const float*  bia = (which == 0) ? bq  : (which == 1) ? bk  : bv;

  const int s0 = blockIdx.x * 256 + w * 32;
  const __bf16* xrow0 = xbf + ((size_t)b * S_ + (s0 + m)) * D_;
  const __bf16* xrow1 = xbf + ((size_t)b * S_ + (s0 + 16 + m)) * D_;
  const __bf16* wth   = wt + (size_t)h * DH_ * D_;

  v8f acc0[4], acc1[4];
#pragma unroll
  for (int t = 0; t < 4; ++t) { acc0[t] = (v8f)0.0f; acc1[t] = (v8f)0.0f; }

  Frag f0, f1;
  load_frag(f0, xrow0, xrow1, wth, m, 0, half);
  for (int k0 = 0; k0 < D_; k0 += 64) {
    load_frag(f1, xrow0, xrow1, wth, m, k0 + 32, half);
    gemm_step(f0, acc0, acc1);
    const int kn = (k0 + 64 < D_) ? (k0 + 64) : k0;  // dummy reload on last
    load_frag(f0, xrow0, xrow1, wth, m, kn, half);
    gemm_step(f1, acc0, acc1);
  }

  const size_t base = (size_t)bh * S_ * DH_;
#pragma unroll
  for (int blk = 0; blk < 2; ++blk) {
#pragma unroll
    for (int t = 0; t < 4; ++t) {
#pragma unroll
      for (int v = 0; v < 8; ++v) {
        float av = (blk == 0) ? acc0[t][v] : acc1[t][v];
        int row = s0 + blk * 16 + v + half * 8;   // C layout: M = v + half*8
        int col = t * 16 + m;                     // C layout: N = lane&15
        __bf16 o = (__bf16)(av + bia[h * DH_ + col]);
        if (which == 2)      vto[base + (size_t)col * S_ + row] = o;  // V^T
        else if (which == 1) ko[base + (size_t)row * DH_ + col] = o;
        else                 qo[base + (size_t)row * DH_ + col] = o;
      }
    }
  }
}

// ---------------------------------------------------------------------------
// K2: flash attention.  One wave owns a 16-query tile; 8 waves share (b,h).
// K/V key-blocks (32 keys) staged to LDS by the TDM (double-buffered, prefetch
// overlapped with compute); one block barrier per key block.  P bounces
// through wave-private LDS with a wave-local DScnt fence only.
__global__ void k_flash(const __bf16* __restrict__ qb,
                        const __bf16* __restrict__ kb_,
                        const __bf16* __restrict__ vtb,
                        __bf16* __restrict__ ctx) {
  __shared__ __align__(16) __bf16 ldsK[2][32 * DH_];   // [buf][key][dh]
  __shared__ __align__(16) __bf16 ldsV[2][DH_ * 32];   // [buf][dh][key]
  __shared__ __align__(16) __bf16 pbuf[8 * 16 * 32];   // wave-private slabs

  const int tid  = threadIdx.x;
  const int w    = tid >> 5;
  const int lane = tid & 31;
  const int m    = lane & 15;
  const int half = lane >> 4;

  const int bh = blockIdx.y;
  const int b  = bh / H_;
  const int h  = bh % H_;
  const int q0 = blockIdx.x * 128 + w * 16;

  const size_t base = (size_t)bh * S_ * DH_;
  const __bf16* kbase = kb_ + base;
  const __bf16* vbase = vtb + base;
  const __bf16* qrow  = qb + base + (size_t)(q0 + m) * DH_;

  v16bf Aq0 = load_a(qrow, 0, half);
  v16bf Aq1 = load_a(qrow, 32, half);

  float mrow[8], lrow[8];
#pragma unroll
  for (int v = 0; v < 8; ++v) { mrow[v] = -3.0e38f; lrow[v] = 0.0f; }
  v8f acc[4];
#pragma unroll
  for (int t = 0; t < 4; ++t) acc[t] = (v8f)0.0f;

  __bf16* pw = &pbuf[w * 16 * 32];
  const float scale = 0.125f;   // 1/sqrt(64)

  // ---- K/V tile fill: TDM (wave 0 issues) or cooperative copy fallback ----
  auto fill = [&](int buf, int kbk) {
#if USE_TDM
    if (w == 0) {
      // K block: 32 consecutive rows x DH = contiguous 4 KB -> 1D tile
      tdm_load(lds_off(ldsK[buf]), kbase + (size_t)kbk * DH_,
               /*tensor_d0=*/32 * DH_, /*tensor_d1=*/1,
               /*tile_d0=*/32 * DH_, /*tile_d1=*/0, /*stride0=*/0);
      // V^T block: 64 dh-rows x 32 keys, row stride S -> 2D tile
      tdm_load(lds_off(ldsV[buf]), vbase + kbk,
               /*tensor_d0=*/S_, /*tensor_d1=*/DH_,
               /*tile_d0=*/32, /*tile_d1=*/DH_, /*stride0=*/S_);
    }
#else
    uint64_t*       lk = (uint64_t*)ldsK[buf];
    const uint64_t* gk = (const uint64_t*)(kbase + (size_t)kbk * DH_);
    for (int i = tid; i < (32 * DH_) / 4; i += 256) lk[i] = gk[i];
    uint64_t* lv = (uint64_t*)ldsV[buf];
    for (int i = tid; i < (DH_ * 32) / 4; i += 256) {
      int r = i >> 3, c = i & 7;
      lv[i] = ((const uint64_t*)(vbase + (size_t)r * S_ + kbk))[c];
    }
#endif
  };

  fill(0, 0);

  int ib = 0;
  for (int kbk = 0; kbk < S_; kbk += 32, ib ^= 1) {
#if USE_TDM
    if (w == 0) __builtin_amdgcn_s_wait_tensorcnt(0);
#endif
    __syncthreads();                       // current K/V buffer visible to all
    if (kbk + 32 < S_) fill(ib ^ 1, kbk + 32);   // prefetch next block

    const __bf16* Kt = ldsK[ib];
    const __bf16* Vt = ldsV[ib];

    // ---- scores: two 16x16 C tiles (keys kbk+n and kbk+16+n) ----
    v8f s0t = (v8f)0.0f, s1t = (v8f)0.0f;
    {
      const __bf16* kr0 = Kt + (size_t)m * DH_;
      const __bf16* kr1 = Kt + (size_t)(16 + m) * DH_;
      v16bf B00 = load_b(kr0, 0, half);
      v16bf B01 = load_b(kr0, 32, half);
      v16bf B10 = load_b(kr1, 0, half);
      v16bf B11 = load_b(kr1, 32, half);
      s0t = wmma_bf16(Aq0, B00, s0t);
      s0t = wmma_bf16(Aq1, B01, s0t);
      s1t = wmma_bf16(Aq0, B10, s1t);
      s1t = wmma_bf16(Aq1, B11, s1t);
    }

    // ---- online softmax (rows = v + half*8, 16 cols across the half) ----
    float f[8];
#pragma unroll
    for (int v = 0; v < 8; ++v) {
      float a0 = s0t[v] * scale, a1 = s1t[v] * scale;
      float rmax = hmax16(fmaxf(a0, a1));
      float nm = fmaxf(mrow[v], rmax);
      f[v] = __expf(mrow[v] - nm);
      mrow[v] = nm;
      float p0 = __expf(a0 - nm), p1 = __expf(a1 - nm);
      s0t[v] = p0; s1t[v] = p1;
      lrow[v] = lrow[v] * f[v] + hsum16(p0 + p1);
    }
#pragma unroll
    for (int t = 0; t < 4; ++t)
#pragma unroll
      for (int v = 0; v < 8; ++v) acc[t][v] *= f[v];

    // ---- P: C-layout -> wave-private LDS [16][32] -> A-layout ----
#pragma unroll
    for (int v = 0; v < 8; ++v) {
      int r = v + half * 8;
      pw[r * 32 + m]      = (__bf16)s0t[v];
      pw[r * 32 + 16 + m] = (__bf16)s1t[v];
    }
    wave_lds_fence();                      // wave-local: DScnt==0 suffices
    v16bf Ap = load_a(pw + m * 32, 0, half);

    // ---- P @ V from LDS (V^T rows are dh, contiguous over keys) ----
#pragma unroll
    for (int t = 0; t < 4; ++t) {
      const __bf16* vr = Vt + (size_t)(t * 16 + m) * 32;
      v16bf Bv = load_b(vr, 0, half);
      acc[t] = wmma_bf16(Ap, Bv, acc[t]);
    }
  }

  // ---- normalize + store context (concat heads -> [B,S,D]) ----
#pragma unroll
  for (int t = 0; t < 4; ++t) {
#pragma unroll
    for (int v = 0; v < 8; ++v) {
      int row = q0 + v + half * 8;
      int col = h * DH_ + t * 16 + m;
      ctx[((size_t)b * S_ + row) * D_ + col] = (__bf16)(acc[t][v] / lrow[v]);
    }
  }
}

// ---------------------------------------------------------------------------
// K3: output projection  out = ctx @ Wo + bo  (Wo pre-transposed [n][k]).
// One wave -> 32 rows x 64 cols, two-phase ping-pong like k_qkv.
__global__ void k_oproj(const __bf16* __restrict__ ctx,
                        const __bf16* __restrict__ wot,
                        const float* __restrict__ bo,
                        float* __restrict__ out) {
  const int w    = threadIdx.x >> 5;
  const int lane = threadIdx.x & 31;
  const int m    = lane & 15;
  const int half = lane >> 4;

  const int row0 = blockIdx.x * 256 + w * 32;
  const int n0   = blockIdx.y * 64;

  const __bf16* crow0 = ctx + (size_t)(row0 + m) * D_;
  const __bf16* crow1 = ctx + (size_t)(row0 + 16 + m) * D_;
  const __bf16* wb    = wot + (size_t)n0 * D_;

  v8f acc0[4], acc1[4];
#pragma unroll
  for (int t = 0; t < 4; ++t) { acc0[t] = (v8f)0.0f; acc1[t] = (v8f)0.0f; }

  Frag f0, f1;
  load_frag(f0, crow0, crow1, wb, m, 0, half);
  for (int k0 = 0; k0 < D_; k0 += 64) {
    load_frag(f1, crow0, crow1, wb, m, k0 + 32, half);
    gemm_step(f0, acc0, acc1);
    const int kn = (k0 + 64 < D_) ? (k0 + 64) : k0;
    load_frag(f0, crow0, crow1, wb, m, kn, half);
    gemm_step(f1, acc0, acc1);
  }

#pragma unroll
  for (int blk = 0; blk < 2; ++blk) {
#pragma unroll
    for (int t = 0; t < 4; ++t) {
#pragma unroll
      for (int v = 0; v < 8; ++v) {
        float av = (blk == 0) ? acc0[t][v] : acc1[t][v];
        int r = row0 + blk * 16 + v + half * 8;
        int c = n0 + t * 16 + m;
        out[(size_t)r * D_ + c] = av + bo[c];
      }
    }
  }
}

// ---------------------------------------------------------------------------
extern "C" void kernel_launch(void* const* d_in, const int* in_sizes, int n_in,
                              void* d_out, int out_size, void* d_ws, size_t ws_size,
                              hipStream_t stream) {
  const float* x  = (const float*)d_in[0];
  const float* Wq = (const float*)d_in[1];
  const float* Wk = (const float*)d_in[2];
  const float* Wv = (const float*)d_in[3];
  const float* bq = (const float*)d_in[4];
  const float* bk = (const float*)d_in[5];
  const float* bv = (const float*)d_in[6];
  const float* Wo = (const float*)d_in[7];
  const float* bo = (const float*)d_in[8];
  float* out = (float*)d_out;

  const size_t N_X  = (size_t)B_ * S_ * D_;        // 4,194,304
  const size_t N_W  = (size_t)H_ * D_ * DH_;       // 1,048,576
  const size_t N_WO = (size_t)D_ * D_;             // 1,048,576
  const size_t N_Q  = (size_t)B_ * H_ * S_ * DH_;  // 4,194,304

  __bf16* ws  = (__bf16*)d_ws;
  __bf16* xbf = ws;
  __bf16* wqt = xbf + N_X;
  __bf16* wkt = wqt + N_W;
  __bf16* wvt = wkt + N_W;
  __bf16* wot = wvt + N_W;
  __bf16* qb  = wot + N_WO;
  __bf16* kb  = qb + N_Q;
  __bf16* vtb = kb + N_Q;
  __bf16* ctx = vtb + N_Q;
  // total ~25.2M bf16 = ~48 MB workspace

  k_cvt<<<2048, 256, 0, stream>>>(x, xbf, (int)N_X);
  k_cvt_tr<<<2048, 256, 0, stream>>>(Wq, wqt, H_, D_, DH_);
  k_cvt_tr<<<2048, 256, 0, stream>>>(Wk, wkt, H_, D_, DH_);
  k_cvt_tr<<<2048, 256, 0, stream>>>(Wv, wvt, H_, D_, DH_);
  k_cvt_tr<<<2048, 256, 0, stream>>>(Wo, wot, 1, D_, D_);

  k_qkv<<<dim3(S_ / 256, B_ * H_ * 3), 256, 0, stream>>>(
      xbf, wqt, wkt, wvt, bq, bk, bv, qb, kb, vtb);

  k_flash<<<dim3(S_ / 128, B_ * H_), 256, 0, stream>>>(qb, kb, vtb, ctx);

  k_oproj<<<dim3((B_ * S_) / 256, D_ / 64), 256, 0, stream>>>(ctx, wot, bo, out);
}
